// FlowRNNCell_14559939133758
// MI455X (gfx1250) — matrix operand order repaired and checked by reading, not compiled, not run on hardware
//
#include <hip/hip_runtime.h>
#include <math.h>

typedef __attribute__((ext_vector_type(16))) __bf16 v16bf;
typedef __attribute__((ext_vector_type(8)))  float  v8f;

#define B_   4
#define C_   64
#define C3_  192
#define H_   128
#define W_   128
#define HW_  16384

__device__ __forceinline__ unsigned short f2bf(float f) {
  unsigned int u = __float_as_uint(f);
  u += 0x7FFFu + ((u >> 16) & 1u);          // round-to-nearest-even
  return (unsigned short)(u >> 16);
}

// K index within a 32-wide WMMA step for A-fragment element e on half-wave `hf`
__device__ __forceinline__ int a_kin(int hf, int e) {
  return 16 * hf + e;
}

// ---------------------------------------------------------------------------
// Weight prep: Wq_eff[m,ci,ki,kj] = sum_o Wk[o,m]*Wq[o,ci,ki,kj], stored bf16
// in WMMA A-fragment order. K-order = (ki*3+kj)*64 + ci. Also zero LN sums.
// ---------------------------------------------------------------------------
__global__ void prep_wq(const float* __restrict__ Wq, const float* __restrict__ Wk,
                        unsigned short* __restrict__ wqarr, float* __restrict__ sums) {
  int gt = blockIdx.x * blockDim.x + threadIdx.x;
  if (gt < 8) sums[gt] = 0.f;
  const int total = 4 * 18 * 32 * 16;
  for (int idx = gt; idx < total; idx += gridDim.x * blockDim.x) {
    int e    = idx & 15;
    int lane = (idx >> 4) & 31;
    int ks   = (idx >> 9) % 18;
    int mt   = (idx >> 9) / 18;
    int m    = mt * 16 + (lane & 15);
    int hf   = lane >> 4;
    int K    = ks * 32 + a_kin(hf, e);
    int r    = K >> 6, ci = K & 63;         // K = r*64 + ci
    int ki   = r / 3,  kj = r % 3;
    float acc = 0.f;
    for (int o = 0; o < 64; ++o)
      acc += Wk[o * 64 + m] * Wq[(size_t)(o * 64 + ci) * 9 + ki * 3 + kj];
    wqarr[idx] = f2bf(acc);
  }
}

// Wg reorder: K-order = (ki*5+kj)*192 + ci
__global__ void prep_wg(const float* __restrict__ Wg, unsigned short* __restrict__ wgarr) {
  int gt = blockIdx.x * blockDim.x + threadIdx.x;
  const int total = 12 * 150 * 32 * 16;
  for (int idx = gt; idx < total; idx += gridDim.x * blockDim.x) {
    int e    = idx & 15;
    int lane = (idx >> 4) & 31;
    int ks   = (idx >> 9) % 150;
    int mt   = (idx >> 9) / 150;
    int m    = mt * 16 + (lane & 15);
    int hf   = lane >> 4;
    int K    = ks * 32 + a_kin(hf, e);
    int r    = K / 192, ci = K % 192;
    int ki   = r / 5,   kj = r % 5;
    wgarr[idx] = f2bf(Wg[((size_t)m * 192 + ci) * 25 + ki * 5 + kj]);
  }
}

// x -> xcat[ch 0..63], f_t -> xcat[ch 64..127] (bf16)
__global__ void convert_kernel(const float* __restrict__ x, const float* __restrict__ ft,
                               unsigned short* __restrict__ xcat) {
  int gt = blockIdx.x * blockDim.x + threadIdx.x;
  const int total = B_ * 128 * HW_;
  for (int idx = gt; idx < total; idx += gridDim.x * blockDim.x) {
    int p  = idx & (HW_ - 1);
    int t  = idx >> 14;
    int ci = t & 127, b = t >> 7;
    float v = (ci < 64) ? x[((size_t)b * C_ + ci) * HW_ + p]
                        : ft[((size_t)b * C_ + (ci - 64)) * HW_ + p];
    xcat[((size_t)b * C3_ + ci) * HW_ + p] = f2bf(v);
  }
}

// ---------------------------------------------------------------------------
// 3x3 conv (f_t * Wq_eff) -> qk, WMMA bf16 implicit GEMM.
// 4 waves/WG, each wave: 16 outch x (4 rows x 16 px), A reused for 4 WMMAs.
// LDS layout channel-blocked: [ci/16][(dy*18+dx)*16 + ci%16].
// ---------------------------------------------------------------------------
__global__ __launch_bounds__(128) void conv3_wmma(
    const unsigned short* __restrict__ xcat,
    const unsigned short* __restrict__ wqarr,
    float* __restrict__ qk) {
  __shared__ alignas(32) unsigned short sB[4 * 6 * 18 * 16];   // 13824 B
  const int b = blockIdx.z, h0 = blockIdx.y * 4, w0 = blockIdx.x * 16;
  const int tid = threadIdx.x;
  const unsigned short* src = xcat + ((size_t)b * C3_ + 64) * HW_;   // f_t channels
  for (int i = tid; i < 6 * 18 * 64; i += 128) {
    int dx = i % 18; int t = i / 18; int dy = t % 6; int ci = t / 6;
    int gh = h0 - 1 + dy, gw = w0 - 1 + dx;
    unsigned short v = 0;                                  // zero pad (SAME)
    if ((unsigned)gh < 128u && (unsigned)gw < 128u)
      v = src[(size_t)ci * HW_ + gh * W_ + gw];
    sB[(ci >> 4) * (6 * 18 * 16) + (dy * 18 + dx) * 16 + (ci & 15)] = v;
  }
  __syncthreads();
  const int wave = tid >> 5, lane = tid & 31;
  const int n = lane & 15, hf = lane >> 4;
  v8f acc[4] = {{}, {}, {}, {}};
  for (int ks = 0; ks < 18; ++ks) {
    int r = ks >> 1;
    int blk = (ks & 1) * 2 + hf;                     // 16-channel block index
    int ki = r / 3, kj = r % 3;
    v16bf a = *reinterpret_cast<const v16bf*>(
        wqarr + ((size_t)(wave * 18 + ks) * 32 + lane) * 16);
    const unsigned short* bbase = &sB[blk * (6 * 18 * 16)];
    #pragma unroll
    for (int rr = 0; rr < 4; ++rr) {
      v16bf bb = *reinterpret_cast<const v16bf*>(
          &bbase[((ki + rr) * 18 + n + kj) * 16]);
      acc[rr] = __builtin_amdgcn_wmma_f32_16x16x32_bf16(false, a, false, bb,
                                                        (short)0, acc[rr], false, false);
    }
  }
  int mrow = hf * 8;
  #pragma unroll
  for (int rr = 0; rr < 4; ++rr) {
    float* dst = qk + ((size_t)b * C_ + wave * 16) * HW_ + (h0 + rr) * W_ + w0 + n;
    #pragma unroll
    for (int r2 = 0; r2 < 8; ++r2)
      dst[(size_t)(mrow + r2) * HW_] = acc[rr][r2];
  }
}

// ---------------------------------------------------------------------------
// Attention: res/prior/softmax/tanh. LDS-tiled x with replicate-pad clamp.
// 16x8 pixel tile per WG, one thread per pixel.
// ---------------------------------------------------------------------------
__global__ __launch_bounds__(128) void attn_kernel(
    const float* __restrict__ x, const float* __restrict__ qk,
    float* __restrict__ ffb, unsigned short* __restrict__ xcat) {
  __shared__ float sX[64 * 20 * 12];            // [ci][dy][dx]
  const int b = blockIdx.z, h0 = blockIdx.y * 16, w0 = blockIdx.x * 8;
  const int tid = threadIdx.x;
  const float* xb = x + (size_t)b * C_ * HW_;
  for (int i = tid; i < 64 * 20 * 12; i += 128) {
    int dx = i % 12; int t = i / 12; int dy = t % 20; int ci = t / 20;
    int gh = h0 - 2 + dy; gh = gh < 0 ? 0 : (gh > 127 ? 127 : gh);  // edge pad
    int gw = w0 - 2 + dx; gw = gw < 0 ? 0 : (gw > 127 ? 127 : gw);
    sX[(ci * 20 + dy) * 12 + dx] = xb[(size_t)ci * HW_ + gh * W_ + gw];
  }
  __syncthreads();
  const int lh = tid >> 3, lw = tid & 7;
  const int h = h0 + lh, w = w0 + lw;
  float s[25];
  #pragma unroll
  for (int k = 0; k < 25; ++k) s[k] = 0.f;
  const float* qkp = qk + (size_t)b * C_ * HW_ + h * W_ + w;
  for (int c = 0; c < 64; ++c) {
    float qkc = qkp[(size_t)c * HW_];
    const float* base = &sX[c * 240];
    float xc = base[(lh + 2) * 12 + lw + 2];
    #pragma unroll
    for (int ki = 0; ki < 5; ++ki)
      #pragma unroll
      for (int kj = 0; kj < 5; ++kj)
        s[ki * 5 + kj] += qkc * (base[(lh + ki) * 12 + lw + kj] - xc);
  }
  float mx = -3.0e38f;
  #pragma unroll
  for (int k = 0; k < 25; ++k) {
    int di = k / 5 - 2, dj = k % 5 - 2;
    float pk = (k == 12) ? 0.f : expf(-(float)(di * di + dj * dj) * (1.f / 16.f));
    s[k] *= pk;
    mx = fmaxf(mx, s[k]);
  }
  float ssum = 0.f;
  #pragma unroll
  for (int k = 0; k < 25; ++k) { s[k] = __expf(s[k] - mx); ssum += s[k]; }
  float inv = 1.f / ssum;
  #pragma unroll
  for (int k = 0; k < 25; ++k) s[k] *= inv;
  float* ffp = ffb + (size_t)b * C_ * HW_ + h * W_ + w;
  unsigned short* xcp = xcat + ((size_t)b * C3_ + 128) * HW_ + h * W_ + w;
  for (int c = 0; c < 64; ++c) {
    const float* base = &sX[c * 240];
    float acc = 0.f;
    #pragma unroll
    for (int ki = 0; ki < 5; ++ki)
      #pragma unroll
      for (int kj = 0; kj < 5; ++kj)
        acc += s[ki * 5 + kj] * base[(lh + ki) * 12 + lw + kj];
    float ffv = tanhf(acc - base[(lh + 2) * 12 + lw + 2]);   // sum(s)=1
    ffp[(size_t)c * HW_] = ffv;
    xcp[(size_t)c * HW_] = f2bf(ffv);
  }
}

// ---------------------------------------------------------------------------
// 5x5 conv (concat * Wg) -> g (+bias), WMMA bf16. 12 waves/WG (one 16-outch
// tile each), 4 output rows per wave so each A fragment feeds 4 WMMAs
// (weight L2 traffic /4). LDS channel-blocked: [ci/16][(dy*20+dx)*16+ci%16].
// Also block-reduces per-batch sum / sum^2 for LayerNorm.
// ---------------------------------------------------------------------------
__global__ __launch_bounds__(384) void conv5_wmma(
    const unsigned short* __restrict__ xcat,
    const unsigned short* __restrict__ wgarr,
    const float* __restrict__ bg,
    float* __restrict__ g, float* __restrict__ sums) {
  __shared__ alignas(32) unsigned short sB[12 * 8 * 20 * 16];  // 61440 B
  __shared__ float rs[384], rq[384];
  const int b = blockIdx.z, h0 = blockIdx.y * 4, w0 = blockIdx.x * 16;
  const int tid = threadIdx.x;
  const unsigned short* src = xcat + (size_t)b * C3_ * HW_;
  for (int i = tid; i < 8 * 20 * 192; i += 384) {
    int dx = i % 20; int t = i / 20; int dy = t % 8; int ci = t / 8;
    int gh = h0 - 2 + dy, gw = w0 - 2 + dx;
    unsigned short v = 0;                                   // zero pad (SAME)
    if ((unsigned)gh < 128u && (unsigned)gw < 128u)
      v = src[(size_t)ci * HW_ + gh * W_ + gw];
    sB[(ci >> 4) * (8 * 20 * 16) + (dy * 20 + dx) * 16 + (ci & 15)] = v;
  }
  __syncthreads();
  const int wave = tid >> 5, lane = tid & 31;
  const int n = lane & 15, hf = lane >> 4;
  v8f acc[4] = {{}, {}, {}, {}};
  for (int ks = 0; ks < 150; ++ks) {
    int r = ks / 6;
    int blk = (ks % 6) * 2 + hf;                     // 16-channel block index
    int ki = r / 5, kj = r % 5;
    v16bf a = *reinterpret_cast<const v16bf*>(
        wgarr + ((size_t)(wave * 150 + ks) * 32 + lane) * 16);
    const unsigned short* bbase = &sB[blk * (8 * 20 * 16)];
    #pragma unroll
    for (int rr = 0; rr < 4; ++rr) {
      v16bf bb = *reinterpret_cast<const v16bf*>(
          &bbase[((ki + rr) * 20 + n + kj) * 16]);
      acc[rr] = __builtin_amdgcn_wmma_f32_16x16x32_bf16(false, a, false, bb,
                                                        (short)0, acc[rr], false, false);
    }
  }
  int mrow = hf * 8;
  float lsum = 0.f, lsq = 0.f;
  #pragma unroll
  for (int rr = 0; rr < 4; ++rr) {
    float* dst = g + ((size_t)b * C3_ + wave * 16) * HW_ + (h0 + rr) * W_ + w0 + n;
    #pragma unroll
    for (int r2 = 0; r2 < 8; ++r2) {
      int oc = wave * 16 + mrow + r2;
      float v = acc[rr][r2] + bg[oc];
      dst[(size_t)(mrow + r2) * HW_] = v;
      lsum += v; lsq += v * v;
    }
  }
  rs[tid] = lsum; rq[tid] = lsq;
  __syncthreads();
  if (tid < 128) { rs[tid] += rs[tid + 256]; rq[tid] += rq[tid + 256]; }
  __syncthreads();
  for (int st = 128; st > 0; st >>= 1) {
    if (tid < st) { rs[tid] += rs[tid + st]; rq[tid] += rq[tid + st]; }
    __syncthreads();
  }
  if (tid == 0) {
    atomicAdd(&sums[2 * b],     rs[0]);
    atomicAdd(&sums[2 * b + 1], rq[0]);
  }
}

// LayerNorm affine + 3-way gate sigmoid + blend
__global__ void finalize_kernel(const float* __restrict__ g, const float* __restrict__ sums,
                                const float* __restrict__ gamma, const float* __restrict__ beta,
                                const float* __restrict__ f_t, const float* __restrict__ ffb,
                                float* __restrict__ out) {
  const float Ninv = 1.f / (float)(C3_ * HW_);
  int gt = blockIdx.x * blockDim.x + threadIdx.x;
  const int total = B_ * C_ * HW_;
  for (int idx = gt; idx < total; idx += gridDim.x * blockDim.x) {
    int p = idx & (HW_ - 1);
    int t = idx >> 14;
    int c = t & 63, b = t >> 6;
    float mu   = sums[2 * b] * Ninv;
    float var  = sums[2 * b + 1] * Ninv - mu * mu;
    float rstd = rsqrtf(var + 1e-5f);
    float z = 0.f;
    #pragma unroll
    for (int i = 0; i < 3; ++i) {
      int ch = c + 64 * i;
      float gv = g[((size_t)b * C3_ + ch) * HW_ + p];
      z += (gv - mu) * rstd * gamma[(size_t)ch * HW_ + p] + beta[(size_t)ch * HW_ + p];
    }
    float gate = 1.f / (1.f + __expf(-z));
    out[idx] = gate * f_t[idx] + (1.f - gate) * ffb[idx];
  }
}

extern "C" void kernel_launch(void* const* d_in, const int* in_sizes, int n_in,
                              void* d_out, int out_size, void* d_ws, size_t ws_size,
                              hipStream_t stream) {
  const float* x     = (const float*)d_in[0];
  const float* f_t   = (const float*)d_in[1];
  const float* Wq    = (const float*)d_in[2];
  const float* Wk    = (const float*)d_in[3];
  const float* Wg    = (const float*)d_in[4];
  const float* bg    = (const float*)d_in[5];
  const float* gamma = (const float*)d_in[6];
  const float* beta  = (const float*)d_in[7];
  float* out = (float*)d_out;

  char* ws = (char*)d_ws;
  size_t off = 0;
  float* sums           = (float*)(ws + off);          off += 256;
  float* qk             = (float*)(ws + off);          off += (size_t)B_ * C_  * HW_ * 4;
  float* ffb            = (float*)(ws + off);          off += (size_t)B_ * C_  * HW_ * 4;
  float* gbuf           = (float*)(ws + off);          off += (size_t)B_ * C3_ * HW_ * 4;
  unsigned short* xcat  = (unsigned short*)(ws + off); off += (size_t)B_ * C3_ * HW_ * 2;
  unsigned short* wqarr = (unsigned short*)(ws + off); off += (size_t)4 * 18 * 32 * 16 * 2;
  unsigned short* wgarr = (unsigned short*)(ws + off); off += (size_t)12 * 150 * 32 * 16 * 2;
  (void)ws_size; (void)in_sizes; (void)n_in; (void)out_size;

  prep_wq<<<64, 256, 0, stream>>>(Wq, Wk, wqarr, sums);
  prep_wg<<<512, 256, 0, stream>>>(Wg, wgarr);
  convert_kernel<<<2048, 256, 0, stream>>>(x, f_t, xcat);
  conv3_wmma<<<dim3(8, 32, 4), 128, 0, stream>>>(xcat, wqarr, qk);
  attn_kernel<<<dim3(16, 8, 4), 128, 0, stream>>>(x, qk, ffb, xcat);
  conv5_wmma<<<dim3(8, 32, 4), 384, 0, stream>>>(xcat, wgarr, bg, gbuf, sums);
  finalize_kernel<<<2048, 256, 0, stream>>>(gbuf, sums, gamma, beta, f_t, ffb, out);
}
